// SparsePairUpdate_3685081940016
// MI455X (gfx1250) — matrix operand order repaired and checked by reading, not compiled, not run on hardware
//
#include <hip/hip_runtime.h>
#include <stdint.h>

#define N_  1024
#define K_  32
#define C_  64
#define CL_ 128
#define H_  128

typedef __attribute__((ext_vector_type(16))) _Float16 v16h;
typedef __attribute__((ext_vector_type(8)))  _Float16 v8h;
typedef __attribute__((ext_vector_type(8)))  float    v8f;
typedef __attribute__((ext_vector_type(4)))  float    v4f;

// ---- WMMA operand loaders, contiguous k order (k = 16h + s in both A and B) ----
// A 16x32 f16: lane L(0-15) row=L holds K 0..15; lane L+16 holds K 16..31.
static __device__ __forceinline__ v16h lda_rm(const _Float16* base, int stride, int kOff, int lane) {
  int r  = lane & 15;
  int hi = lane >> 4;
  const _Float16* p = base + r * stride + kOff + hi * 16;
  v8h x = *(const v8h*)p;
  v8h y = *(const v8h*)(p + 8);
  v16h a;
#pragma unroll
  for (int i = 0; i < 8; ++i) { a[i] = x[i]; a[i + 8] = y[i]; }
  return a;
}
// A-load forming (b0 + b1) on the fly (layer-1 npair = left + right_gathered), kOff = 0.
static __device__ __forceinline__ v16h lda_rm_sum(const _Float16* b0, const _Float16* b1, int stride, int lane) {
  int r  = lane & 15;
  int hi = lane >> 4;
  int o  = r * stride + hi * 16;
  v8h x = *(const v8h*)(b0 + o)      + *(const v8h*)(b1 + o);
  v8h y = *(const v8h*)(b0 + o + 8) + *(const v8h*)(b1 + o + 8);
  v16h a;
#pragma unroll
  for (int i = 0; i < 8; ++i) { a[i] = x[i]; a[i + 8] = y[i]; }
  return a;
}
// B 32x16 f16: lane n(0-15) col=n holds K 0..15; lane n+16 holds K 16..31 (column-major LDS).
static __device__ __forceinline__ v16h ldb_cm(const _Float16* base, int stride, int colOff, int kOff, int lane) {
  int c  = colOff + (lane & 15);
  int hi = lane >> 4;
  return *(const v16h*)(base + c * stride + kOff + hi * 16);
}
static __device__ __forceinline__ v8f wmma_f16(v16h a, v16h b, v8f c) {
  return __builtin_amdgcn_wmma_f32_16x16x32_f16(false, a, false, b, (short)0, c, false, false);
}
static __device__ __forceinline__ float gelu_tanh(float x) {
  const float c0 = 0.7978845608028654f;  // sqrt(2/pi)
  return 0.5f * x * (1.f + tanhf(c0 * (x + 0.044715f * x * x * x)));
}
// Async gather of a 2KB tile (K_*32 halfs) into LDS: 32 lanes x 4 x 16B, ASYNCcnt-tracked.
static __device__ __forceinline__ void async_fetch_tile(const _Float16* gsrc, _Float16* ldst, int lane) {
  unsigned lds = (unsigned)(uintptr_t)(void*)ldst + (unsigned)lane * 16u;
  uint64_t ga  = (uint64_t)(uintptr_t)(const void*)gsrc + (uint64_t)lane * 16u;
#pragma unroll
  for (int q = 0; q < 4; ++q) {
    unsigned l = lds + (unsigned)q * 512u;
    uint64_t g = ga + (uint64_t)q * 512u;
    asm volatile("global_load_async_to_lds_b128 %0, %1, off" :: "v"(l), "v"(g) : "memory");
  }
}

// ---------------- kernel: f32 -> f16 weight conversion ----------------
__global__ __launch_bounds__(256) void k_cvt(const float* __restrict__ src, _Float16* __restrict__ dst, int n) {
  int i = blockIdx.x * 256 + threadIdx.x;
  if (i < n) dst[i] = (_Float16)src[i];
}

// ---------------- kernel A: la = local@Wa, lb = local@Wb, inter = local@W_int + b_int ----------------
__global__ __launch_bounds__(256) void k_local_proj(
    const float* __restrict__ local,
    const _Float16* __restrict__ Wa16, const _Float16* __restrict__ Wb16,
    const _Float16* __restrict__ Wint16, const float* __restrict__ b_int,
    float* __restrict__ la, float* __restrict__ lb, float* __restrict__ inter) {
  __shared__ alignas(32) _Float16 sA[16 * CL_];
  __shared__ alignas(32) _Float16 sW[3][C_ * CL_];  // column-major [64 cols][128 k]
  int tid = threadIdx.x;
  int m0 = blockIdx.x * 16;
  for (int i = tid; i < 16 * CL_; i += 256) {
    int r = i >> 7, k = i & 127;
    sA[i] = (_Float16)local[(size_t)(m0 + r) * CL_ + k];
  }
  for (int i = tid; i < C_ * CL_; i += 256) {
    int col = i >> 7, k = i & 127;
    sW[0][i] = Wa16[k * C_ + col];
    sW[1][i] = Wb16[k * C_ + col];
    sW[2][i] = Wint16[k * C_ + col];
  }
  __syncthreads();
  int wave = tid >> 5, lane = tid & 31;
  for (int t = wave; t < 12; t += 8) {
    int w = t >> 2;   // which output
    int nt = t & 3;   // 16-col tile
    v8f acc = {};
#pragma unroll
    for (int ks = 0; ks < 4; ++ks) {
      v16h a = lda_rm(sA, CL_, ks * 32, lane);
      v16h b = ldb_cm(sW[w], CL_, nt * 16, ks * 32, lane);
      acc = wmma_f16(a, b, acc);
    }
    float* outp = (w == 0) ? la : (w == 1) ? lb : inter;
    int hi = lane >> 4, col = nt * 16 + (lane & 15);
    float bias = (w == 2) ? b_int[col] : 0.f;
#pragma unroll
    for (int i = 0; i < 8; ++i) {
      int m = i + 8 * hi;
      outp[(size_t)(m0 + m) * C_ + col] = acc[i] + bias;
    }
  }
}

// ---------------- kernel B: per-edge features (LN gather, pu@W_aug, lp@{W_lin,W_left,W_right}) ----------------
__global__ __launch_bounds__(256) void k_edge(
    const float* __restrict__ pair, const float* __restrict__ pair_update,
    const int* __restrict__ nbr,
    const float* __restrict__ ln_g, const float* __restrict__ ln_b,
    const _Float16* __restrict__ Waug16, const _Float16* __restrict__ Wlin16,
    const _Float16* __restrict__ Wl16, const _Float16* __restrict__ Wr16,
    const float* __restrict__ b_left, const float* __restrict__ b_right,
    const float* __restrict__ la, const float* __restrict__ lb,
    float* __restrict__ linear, _Float16* __restrict__ left16, _Float16* __restrict__ right16) {
  __shared__ alignas(32) _Float16 sWaug[C_ * C_];  // col-major [64][64]
  __shared__ alignas(32) _Float16 sWlin[C_ * C_];
  __shared__ alignas(32) _Float16 sWl[32 * C_];    // col-major [32][64]
  __shared__ alignas(32) _Float16 sWr[32 * C_];
  __shared__ alignas(32) _Float16 sPu[16 * C_];
  __shared__ alignas(32) _Float16 sLp[16 * C_];
  __shared__ float sLn[16 * C_];
  __shared__ int sJn[16];

  int tid = threadIdx.x;
  int e0 = blockIdx.x * 16;
  int n = e0 >> 5;  // K_ == 32 -> all 16 edges share row n
  for (int i = tid; i < C_ * C_; i += 256) {
    int col = i >> 6, k = i & 63;
    sWaug[i] = Waug16[k * C_ + col];
    sWlin[i] = Wlin16[k * C_ + col];
  }
  for (int i = tid; i < 32 * C_; i += 256) {
    int col = i >> 6, k = i & 63;
    sWl[i] = Wl16[k * 32 + col];
    sWr[i] = Wr16[k * 32 + col];
  }
  if (tid < 16) {
    int j = nbr[e0 + tid];
    sJn[tid] = (j < 0) ? j + N_ : j;  // JAX negative-index wrap
  }
  __syncthreads();

  // LayerNorm of gathered pair rows: 16 rows x 16 lanes, 4 elems each
  {
    int r = tid >> 4, lr = tid & 15;
    int jn = sJn[r];
    const float* src = pair + ((size_t)n * N_ + jn) * C_;
    float x[4];
    float s = 0.f;
#pragma unroll
    for (int q = 0; q < 4; ++q) { x[q] = src[lr * 4 + q]; s += x[q]; }
#pragma unroll
    for (int off = 8; off; off >>= 1) s += __shfl_xor(s, off, 16);
    float mean = s * (1.f / 64.f);
    float v = 0.f;
#pragma unroll
    for (int q = 0; q < 4; ++q) { float d = x[q] - mean; v += d * d; }
#pragma unroll
    for (int off = 8; off; off >>= 1) v += __shfl_xor(v, off, 16);
    float rstd = rsqrtf(v * (1.f / 64.f) + 1e-5f);
#pragma unroll
    for (int q = 0; q < 4; ++q) {
      int c = lr * 4 + q;
      sLn[r * C_ + c] = (x[q] - mean) * rstd * ln_g[c] + ln_b[c];
    }
  }
  // pu = pair_update + la[n] + lb[jn]
  for (int i = tid; i < 16 * C_; i += 256) {
    int r = i >> 6, c = i & 63;
    int jn = sJn[r];
    float v = pair_update[(size_t)(e0 + r) * C_ + c] + la[(size_t)n * C_ + c] + lb[(size_t)jn * C_ + c];
    sPu[i] = (_Float16)v;
  }
  __syncthreads();

  int wave = tid >> 5, lane = tid & 31;
  // lp = LN + pu @ W_aug  (waves 0-3, one 16-col tile each, K=64 -> 2 WMMA)
  if (wave < 4) {
    v8f acc = {};
#pragma unroll
    for (int ks = 0; ks < 2; ++ks) {
      v16h a = lda_rm(sPu, C_, ks * 32, lane);
      v16h b = ldb_cm(sWaug, C_, wave * 16, ks * 32, lane);
      acc = wmma_f16(a, b, acc);
    }
    int hi = lane >> 4, col = wave * 16 + (lane & 15);
#pragma unroll
    for (int i = 0; i < 8; ++i) {
      int m = i + 8 * hi;
      sLp[m * C_ + col] = (_Float16)(sLn[m * C_ + col] + acc[i]);
    }
  }
  __syncthreads();

  // waves 0-3: linear = lp@W_lin ; waves 4-5: left ; waves 6-7: right
  if (wave < 4) {
    v8f acc = {};
#pragma unroll
    for (int ks = 0; ks < 2; ++ks) {
      v16h a = lda_rm(sLp, C_, ks * 32, lane);
      v16h b = ldb_cm(sWlin, C_, wave * 16, ks * 32, lane);
      acc = wmma_f16(a, b, acc);
    }
    int hi = lane >> 4, col = wave * 16 + (lane & 15);
#pragma unroll
    for (int i = 0; i < 8; ++i) {
      int m = i + 8 * hi;
      linear[(size_t)(e0 + m) * C_ + col] = acc[i];
    }
  } else {
    int half = wave - 4;
    const _Float16* W = (half < 2) ? sWl : sWr;
    const float* bias = (half < 2) ? b_left : b_right;
    _Float16* dst = (half < 2) ? left16 : right16;
    int nt = half & 1;
    v8f acc = {};
#pragma unroll
    for (int ks = 0; ks < 2; ++ks) {
      v16h a = lda_rm(sLp, C_, ks * 32, lane);
      v16h b = ldb_cm(W, C_, nt * 16, ks * 32, lane);
      acc = wmma_f16(a, b, acc);
    }
    int hi = lane >> 4, col = nt * 16 + (lane & 15);
#pragma unroll
    for (int i = 0; i < 8; ++i) {
      int m = i + 8 * hi;
      dst[(size_t)(e0 + m) * 32 + col] = (_Float16)(acc[i] + bias[col]);
    }
  }
}

// ---------------- kernel C: row copy (NT) + fused edge-pair MLP (WMMA, async-LDS prefetch)
//                  + masked mean + scatter-add ----------------
__global__ __launch_bounds__(256) void k_mlp_scatter(
    const float* __restrict__ pair,
    const _Float16* __restrict__ left16, const _Float16* __restrict__ right16,
    const int* __restrict__ nbr, const uint8_t* __restrict__ mask,
    const _Float16* __restrict__ W1_16, const _Float16* __restrict__ W2_16,
    const float* __restrict__ inter, const float* __restrict__ linear,
    float* __restrict__ out) {
  __shared__ alignas(32) _Float16 sW1[H_ * 32];   // col-major [128 cols][32 k]
  __shared__ alignas(32) _Float16 sW2[C_ * H_];   // col-major [64 cols][128 k]
  __shared__ alignas(32) _Float16 sLeft[K_ * 32];
  __shared__ alignas(32) _Float16 sRg[2][K_ * 32];  // double-buffered async right-gather
  __shared__ alignas(32) _Float16 sHid[K_ * H_];
  __shared__ int sJn[K_];
  __shared__ int sNb[K_];
  __shared__ unsigned sPm[K_];
  __shared__ int sCnt[K_];
  __shared__ int sActive[K_];
  __shared__ int sNact;

  int n = blockIdx.x;
  int tid = threadIdx.x;
  int wave = tid >> 5, lane = tid & 31;

  // ---- stream row n of pair -> out with non-temporal b128 (one-shot 512MB stream; keep out of L2)
  {
    const v4f* src = (const v4f*)(pair + (size_t)n * N_ * C_);
    v4f* dst = (v4f*)(out + (size_t)n * N_ * C_);
    for (int i = tid; i < (N_ * C_) / 4; i += 256) {
      v4f t = __builtin_nontemporal_load(src + i);
      __builtin_nontemporal_store(t, dst + i);
    }
    __threadfence();  // order copy stores before this block's later atomics on the same row
  }

  for (int i = tid; i < H_ * 32; i += 256) {
    int col = i >> 5, k = i & 31;
    sW1[i] = W1_16[k * H_ + col];
  }
  for (int i = tid; i < C_ * H_; i += 256) {
    int col = i >> 7, k = i & 127;
    sW2[i] = W2_16[k * C_ + col];
  }
  for (int i = tid; i < K_ * 32; i += 256) sLeft[i] = left16[(size_t)n * K_ * 32 + i];
  if (tid < K_) {
    int j = nbr[n * K_ + tid];
    sNb[tid] = j;
    sJn[tid] = (j < 0) ? j + N_ : j;
  }
  __syncthreads();

  // pm[j][k] = (nbr[n,j]==-1) && (pn != -1) && mask[wrap(pn)] && (nbr[n,k]!=-1),
  // pn = nbr[wrap(nbr[n,j]), k]   (matches the reference's where() exactly)
  if (tid < K_) {
    int j = sNb[tid], jw = sJn[tid];
    unsigned pmm = 0u;
    for (int k = 0; k < K_; ++k) {
      int pn = (j == -1) ? nbr[jw * K_ + k] : -1;
      int ok = 0;
      if (pn != -1) {
        int pw = (pn < 0) ? pn + N_ : pn;
        ok = (mask[pw] != 0) && (sNb[k] != -1);
      }
      pmm |= (unsigned)ok << k;
    }
    sPm[tid] = pmm;
  }
  __syncthreads();
  if (tid < K_) {
    int c = 0;
    for (int j = 0; j < K_; ++j) c += (sPm[j] >> tid) & 1u;
    sCnt[tid] = c;
  }
  if (tid == 0) {
    int c = 0;
    for (int j = 0; j < K_; ++j)
      if (sPm[j]) sActive[c++] = j;   // skip j's with all-zero pm: removes ~80% of MLP FLOPs exactly
    sNact = c;
  }
  __syncthreads();
  int nact = sNact;

  // prime the async pipeline for the first active j
  if (nact > 0 && wave == 0)
    async_fetch_tile(right16 + (size_t)sJn[sActive[0]] * K_ * 32, sRg[0], lane);

  int kt = wave & 1, ct = wave >> 1;  // wave-owned acc tile: rows kt*16.., cols ct*16..
  v8f acc = {};
  for (int a = 0; a < nact; ++a) {
    int j = sActive[a];
    int p = a & 1;
    __syncthreads();  // prev layer2 done (sHid free); prefetch target sRg[1-p] free
    if (wave == 0) {
      asm volatile("s_wait_asynccnt 0x0" ::: "memory");  // tile a landed in LDS
      if (a + 1 < nact)
        async_fetch_tile(right16 + (size_t)sJn[sActive[a + 1]] * K_ * 32, sRg[(a + 1) & 1], lane);
    }
    __syncthreads();  // sRg[p] visible to all waves
    // layer1: (left + rg) [32x32] @ W1 [32x128]; 16 tiles over 8 waves, K=32 -> 1 WMMA each
    for (int t = wave; t < 16; t += 8) {
      int mt = t & 1, nt = t >> 1;
      v16h av = lda_rm_sum(sLeft + mt * 16 * 32, sRg[p] + mt * 16 * 32, 32, lane);
      v16h bv = ldb_cm(sW1, 32, nt * 16, 0, lane);
      v8f h = {};
      h = wmma_f16(av, bv, h);
      int hi = lane >> 4, col = nt * 16 + (lane & 15);
#pragma unroll
      for (int i = 0; i < 8; ++i) {
        int m = mt * 16 + i + 8 * hi;
        sHid[m * H_ + col] = (_Float16)gelu_tanh(h[i]);
      }
    }
    __syncthreads();
    // layer2: gelu(h) [32x128] @ W2 [128x64]; wave tile (kt, ct), K=128 -> 4 WMMA
    v8f d = {};
#pragma unroll
    for (int ks = 0; ks < 4; ++ks) {
      v16h av = lda_rm(sHid + kt * 16 * H_, H_, ks * 32, lane);
      v16h bv = ldb_cm(sW2, H_, ct * 16, ks * 32, lane);
      d = wmma_f16(av, bv, d);
    }
    unsigned pmm = sPm[j];
    int hi = lane >> 4;
#pragma unroll
    for (int i = 0; i < 8; ++i) {
      int k = kt * 16 + i + 8 * hi;
      if ((pmm >> k) & 1u) acc[i] += d[i];
    }
  }

  // finalize: acc/cnt + inter + linear, scatter-add to out[n, wrap(nbr[n,k]), :]
  __syncthreads();
  int hi = lane >> 4, col = ct * 16 + (lane & 15);
  float interv = inter[(size_t)n * C_ + col];
#pragma unroll
  for (int i = 0; i < 8; ++i) {
    int k = kt * 16 + i + 8 * hi;
    float cf = (sCnt[k] > 1) ? (float)sCnt[k] : 1.f;
    float val = acc[i] / cf + interv + linear[(size_t)(n * K_ + k) * C_ + col];
    int tr = sJn[k];
    atomicAdd(&out[((size_t)n * N_ + tr) * C_ + col], val);
  }
}

// ---------------- host ----------------
extern "C" void kernel_launch(void* const* d_in, const int* in_sizes, int n_in,
                              void* d_out, int out_size, void* d_ws, size_t ws_size,
                              hipStream_t stream) {
  (void)in_sizes; (void)n_in; (void)out_size; (void)ws_size;
  const float* local       = (const float*)d_in[0];
  const float* pair        = (const float*)d_in[1];
  const float* pair_update = (const float*)d_in[2];
  const int*   neighbours  = (const int*)d_in[3];
  const uint8_t* mask      = (const uint8_t*)d_in[4];
  const float* Wa      = (const float*)d_in[5];
  const float* Wb      = (const float*)d_in[6];
  const float* ln_g    = (const float*)d_in[7];
  const float* ln_b    = (const float*)d_in[8];
  const float* W_aug   = (const float*)d_in[9];
  const float* W_lin   = (const float*)d_in[10];
  const float* W_left  = (const float*)d_in[11];
  const float* b_left  = (const float*)d_in[12];
  const float* W_right = (const float*)d_in[13];
  const float* b_right = (const float*)d_in[14];
  const float* W1      = (const float*)d_in[15];
  const float* W2      = (const float*)d_in[16];
  const float* W_int   = (const float*)d_in[17];
  const float* b_int   = (const float*)d_in[18];

  char* ws = (char*)d_ws;
  float*    la      = (float*)(ws + 0);
  float*    lb      = (float*)(ws + 262144);
  float*    inter   = (float*)(ws + 524288);
  float*    linear  = (float*)(ws + 786432);          // 8 MB
  _Float16* left16  = (_Float16*)(ws + 9175040);      // 2 MB
  _Float16* right16 = (_Float16*)(ws + 11272192);     // 2 MB
  _Float16* Wa16    = (_Float16*)(ws + 13369344);
  _Float16* Wb16    = (_Float16*)(ws + 13385728);
  _Float16* Wint16  = (_Float16*)(ws + 13402112);
  _Float16* Waug16  = (_Float16*)(ws + 13418496);
  _Float16* Wlin16  = (_Float16*)(ws + 13426688);
  _Float16* Wl16    = (_Float16*)(ws + 13434880);
  _Float16* Wr16    = (_Float16*)(ws + 13438976);
  _Float16* W1_16   = (_Float16*)(ws + 13443072);
  _Float16* W2_16   = (_Float16*)(ws + 13451264);

  struct { const float* s; _Float16* d; int n; } cv[9] = {
      {Wa, Wa16, CL_ * C_}, {Wb, Wb16, CL_ * C_}, {W_int, Wint16, CL_ * C_},
      {W_aug, Waug16, C_ * C_}, {W_lin, Wlin16, C_ * C_},
      {W_left, Wl16, C_ * 32}, {W_right, Wr16, C_ * 32},
      {W1, W1_16, 32 * H_}, {W2, W2_16, H_ * C_}};
  for (int i = 0; i < 9; ++i)
    k_cvt<<<(cv[i].n + 255) / 256, 256, 0, stream>>>(cv[i].s, cv[i].d, cv[i].n);

  k_local_proj<<<N_ / 16, 256, 0, stream>>>(local, Wa16, Wb16, Wint16, b_int, la, lb, inter);

  k_edge<<<(N_ * K_) / 16, 256, 0, stream>>>(pair, pair_update, neighbours, ln_g, ln_b,
                                             Waug16, Wlin16, Wl16, Wr16, b_left, b_right,
                                             la, lb, linear, left16, right16);

  k_mlp_scatter<<<N_, 256, 0, stream>>>(pair, left16, right16, neighbours, mask, W1_16, W2_16,
                                        inter, linear, (float*)d_out);
}
